// Encoder_GIN_8770323218938
// MI455X (gfx1250) — matrix operand order repaired and checked by reading, not compiled, not run on hardware
//
#include <hip/hip_runtime.h>

#define N_NODES    50000
#define N_EDGES    800000
#define DIM        128
#define N_LAYERS   3
#define NUM_GRAPHS 500
#define OUT_STRIDE (N_LAYERS * DIM)                 // 384
#define GRAPH_EMB_SIZE (NUM_GRAPHS * OUT_STRIDE)    // 192000
#define N_TILES    (N_NODES / 16)                   // 3125 (exact)
#define MLP_WAVES  4
#define WFRAG_PER_MAT (8 * 4 * 32 * 16)             // 16384 halfs per 128x128 matrix

typedef __attribute__((ext_vector_type(16))) _Float16 v16h;
typedef __attribute__((ext_vector_type(8)))  float    v8f;

// Guaranteed single-instruction HW fp32 atomic, device scope (cross-WGP
// reduction => RMW must happen at a cache level >= DEV per CDNA5 scope rules).
// Non-returning form (TH default) -> tracked with STOREcnt, fire-and-forget.
__device__ __forceinline__ void atom_add_f32(float* p, float v) {
    asm volatile("global_atomic_add_f32 %0, %1, off scope:SCOPE_DEV"
                 :: "v"(p), "v"(v) : "memory");
}

// ---------------------------------------------------------------------------
// Zero the graph-embedding region of d_out (pool uses atomics; the harness
// does not re-poison between replays, so this must run every call).
__global__ void zero_kernel(float* __restrict__ p, int n) {
    int i = blockIdx.x * blockDim.x + threadIdx.x;
    if (i < n) p[i] = 0.0f;
}

// ---------------------------------------------------------------------------
// agg = h  (the (1+eps)*h_i term, eps=0). h stride 128 (x) or 384 (node_cat).
__global__ void init_agg_kernel(const float* __restrict__ h, int hstride,
                                float* __restrict__ agg) {
    int i = blockIdx.x * blockDim.x + threadIdx.x;
    if (i >= N_NODES * DIM) return;
    int v = i >> 7, c = i & (DIM - 1);
    agg[i] = h[(size_t)v * hstride + c];
}

// ---------------------------------------------------------------------------
// One wave (32 lanes) per edge: lane gathers float4 of h[src], does 4 f32
// hardware atomic adds into agg[dst]. agg (25.6 MB) is L2-resident (192 MB L2)
// so the atomics execute at L2 without HBM round trips.
__global__ void scatter_kernel(const float* __restrict__ h, int hstride,
                               const int* __restrict__ src,
                               const int* __restrict__ dst,
                               float* __restrict__ agg) {
    unsigned t = blockIdx.x * blockDim.x + threadIdx.x;
    unsigned e = t >> 5;
    if (e >= N_EDGES) return;
    int lane = t & 31;
    int s = src[e], d = dst[e];
    float4 v = ((const float4*)(h + (size_t)s * hstride))[lane];
    float* ap = agg + (size_t)d * DIM + lane * 4;
    atom_add_f32(ap + 0, v.x);
    atom_add_f32(ap + 1, v.y);
    atom_add_f32(ap + 2, v.z);
    atom_add_f32(ap + 3, v.w);
}

// ---------------------------------------------------------------------------
// Pre-swizzle the 6 weight matrices (f32 row-major 128x128) into wave32 WMMA
// B-fragment order, converted to f16. Fragment (nb,kb): lane<16 holds column
// N=nb*16+lane, K=kb*32+{0..7,16..23} (2 packed f16 per VGPR); lane>=16 holds
// K=kb*32+{8..15,24..31}. Stored so a wave reads a fragment as one contiguous
// 32B-per-lane load.
__global__ void prep_wfrag_kernel(const float* __restrict__ W1,
                                  const float* __restrict__ W2,
                                  _Float16* __restrict__ wf) {
    int idx = blockIdx.x * blockDim.x + threadIdx.x;
    if (idx >= N_LAYERS * 2 * 8 * 4 * 32) return;
    int lane  = idx & 31;  int t = idx >> 5;
    int kb    = t & 3;     t >>= 2;
    int nb    = t & 7;     t >>= 3;
    int which = t & 1;     int layer = t >> 1;
    const float* W = (which ? W2 : W1) + (size_t)layer * DIM * DIM;
    int n     = nb * 16 + (lane & 15);
    int kbase = kb * 32 + ((lane >> 4) ? 8 : 0);
    _Float16* dstp = wf + ((size_t)((layer * 2 + which) * 32 + nb * 4 + kb) * 32 + lane) * 16;
#pragma unroll
    for (int j = 0; j < 16; ++j)
        dstp[j] = (_Float16)W[(size_t)(kbase + j + (j >= 8 ? 8 : 0)) * DIM + n];
}

// ---------------------------------------------------------------------------
// Load a 16x32 f16 A-fragment from a row-major f32 row pointer.
// A layout (wave32): lane<16 is row M=lane, holding K = {c0..c0+7, c0+16..c0+23}
// with c0 = kb*32; lane>=16 same rows, K offset +8 (i.e. {8..15, 24..31}).
__device__ __forceinline__ v16h load_a_frag(const float* __restrict__ rowp, int c0) {
    float4 f0 = *(const float4*)(rowp + c0);
    float4 f1 = *(const float4*)(rowp + c0 + 4);
    float4 f2 = *(const float4*)(rowp + c0 + 16);
    float4 f3 = *(const float4*)(rowp + c0 + 20);
    v16h a;
    a[0]  = (_Float16)f0.x; a[1]  = (_Float16)f0.y;
    a[2]  = (_Float16)f0.z; a[3]  = (_Float16)f0.w;
    a[4]  = (_Float16)f1.x; a[5]  = (_Float16)f1.y;
    a[6]  = (_Float16)f1.z; a[7]  = (_Float16)f1.w;
    a[8]  = (_Float16)f2.x; a[9]  = (_Float16)f2.y;
    a[10] = (_Float16)f2.z; a[11] = (_Float16)f2.w;
    a[12] = (_Float16)f3.x; a[13] = (_Float16)f3.y;
    a[14] = (_Float16)f3.z; a[15] = (_Float16)f3.w;
    return a;
}

// ---------------------------------------------------------------------------
// Fused GIN MLP: Y = relu(relu(A@W1 + b1)@W2 + b2), A = agg (50000x128).
// One wave per 16-row tile; per stage: 8 N-blocks x 4 K-blocks of
// v_wmma_f32_16x16x32_f16 (f32 accumulate). Intermediate 16x128 tile staged
// in per-wave LDS (stride 132 to dodge bank aliasing) to convert the C/D
// layout into A-fragment layout for stage 2. Output written into the
// node_cat region of d_out with row stride 384.
__launch_bounds__(MLP_WAVES * 32)
__global__ void gin_mlp_wmma_kernel(const float* __restrict__ agg,
                                    const _Float16* __restrict__ wf1,
                                    const _Float16* __restrict__ wf2,
                                    const float* __restrict__ b1,
                                    const float* __restrict__ b2,
                                    float* __restrict__ out) {
    __shared__ float T[MLP_WAVES][16 * 132];
    const int wave = threadIdx.x >> 5;
    const int lane = threadIdx.x & 31;
    const int tile = blockIdx.x * MLP_WAVES + wave;
    if (tile >= N_TILES) return;           // wave-uniform: EXEC stays all-1s
    const int row0 = tile << 4;
    const int m    = lane & 15;
    const int half = lane >> 4;

    // ---- stage 1: T = relu(A @ W1 + b1) ----
    v16h a[4];
    const float* arow = agg + (size_t)(row0 + m) * DIM;
#pragma unroll
    for (int kb = 0; kb < 4; ++kb)
        a[kb] = load_a_frag(arow, kb * 32 + half * 8);

    float* Tw = &T[wave][0];
#pragma unroll
    for (int nb = 0; nb < 8; ++nb) {
        v8f c = {};
#pragma unroll
        for (int kb = 0; kb < 4; ++kb) {
            v16h b = *(const v16h*)(wf1 + (size_t)((nb * 4 + kb) * 32 + lane) * 16);
            c = __builtin_amdgcn_wmma_f32_16x16x32_f16(false, a[kb], false, b,
                                                       (short)0, c, false, false);
        }
        const int n = nb * 16 + m;
        const float bias = b1[n];
#pragma unroll
        for (int r = 0; r < 8; ++r) {
            float v = c[r] + bias;
            Tw[(r + half * 8) * 132 + n] = v > 0.0f ? v : 0.0f;   // C/D: M=r+8*half, N=n
        }
    }
    // Same-wave LDS RAW across lanes: wait for DS stores before re-reading.
    asm volatile("s_wait_dscnt 0x0" ::: "memory");

    // ---- stage 2: Y = relu(T @ W2 + b2) ----
    v16h a2[4];
    const float* trow = Tw + m * 132;
#pragma unroll
    for (int kb = 0; kb < 4; ++kb)
        a2[kb] = load_a_frag(trow, kb * 32 + half * 8);

#pragma unroll
    for (int nb = 0; nb < 8; ++nb) {
        v8f c = {};
#pragma unroll
        for (int kb = 0; kb < 4; ++kb) {
            v16h b = *(const v16h*)(wf2 + (size_t)((nb * 4 + kb) * 32 + lane) * 16);
            c = __builtin_amdgcn_wmma_f32_16x16x32_f16(false, a2[kb], false, b,
                                                       (short)0, c, false, false);
        }
        const int n = nb * 16 + m;
        const float bias = b2[n];
#pragma unroll
        for (int r = 0; r < 8; ++r) {
            float v = c[r] + bias;
            out[(size_t)(row0 + r + half * 8) * OUT_STRIDE + n] = v > 0.0f ? v : 0.0f;
        }
    }
}

// ---------------------------------------------------------------------------
// global_add_pool: one float4 per thread, 4 HW f32 atomics into the 768 KB
// (L2-resident) graph_emb.
__global__ void pool_kernel(const float* __restrict__ node_cat,
                            const int* __restrict__ batch,
                            float* __restrict__ graph_emb) {
    unsigned i = blockIdx.x * blockDim.x + threadIdx.x;
    if (i >= (unsigned)N_NODES * (OUT_STRIDE / 4)) return;
    unsigned v  = i / (OUT_STRIDE / 4);
    unsigned c4 = i - v * (OUT_STRIDE / 4);
    int g = batch[v];
    float4 val = ((const float4*)(node_cat + (size_t)v * OUT_STRIDE))[c4];
    float* gp = graph_emb + (size_t)g * OUT_STRIDE + c4 * 4;
    atom_add_f32(gp + 0, val.x);
    atom_add_f32(gp + 1, val.y);
    atom_add_f32(gp + 2, val.z);
    atom_add_f32(gp + 3, val.w);
}

// ---------------------------------------------------------------------------
extern "C" void kernel_launch(void* const* d_in, const int* in_sizes, int n_in,
                              void* d_out, int out_size, void* d_ws, size_t ws_size,
                              hipStream_t stream) {
    (void)in_sizes; (void)n_in; (void)out_size; (void)ws_size;

    const float* x     = (const float*)d_in[0];
    const int*   edge  = (const int*)d_in[1];     // [2, N_EDGES]
    const int*   batch = (const int*)d_in[2];
    const float* W1    = (const float*)d_in[3];   // [3,128,128]
    const float* b1    = (const float*)d_in[4];   // [3,128]
    const float* W2    = (const float*)d_in[5];
    const float* b2    = (const float*)d_in[6];
    const int* src = edge;
    const int* dst = edge + N_EDGES;

    float* graph_emb = (float*)d_out;                    // [500, 384]
    float* node_cat  = (float*)d_out + GRAPH_EMB_SIZE;   // [50000, 384]

    float*    agg   = (float*)d_ws;                               // 25.6 MB
    _Float16* wfrag = (_Float16*)((char*)d_ws +
                                  (size_t)N_NODES * DIM * sizeof(float));

    zero_kernel<<<(GRAPH_EMB_SIZE + 255) / 256, 256, 0, stream>>>(graph_emb, GRAPH_EMB_SIZE);
    prep_wfrag_kernel<<<(N_LAYERS * 2 * 8 * 4 * 32 + 255) / 256, 256, 0, stream>>>(W1, W2, wfrag);

    const float* h = x;
    int hstride = DIM;
    for (int l = 0; l < N_LAYERS; ++l) {
        init_agg_kernel<<<(N_NODES * DIM + 255) / 256, 256, 0, stream>>>(h, hstride, agg);
        scatter_kernel<<<(N_EDGES * 32 + 255) / 256, 256, 0, stream>>>(h, hstride, src, dst, agg);
        gin_mlp_wmma_kernel<<<(N_TILES + MLP_WAVES - 1) / MLP_WAVES, MLP_WAVES * 32, 0, stream>>>(
            agg,
            wfrag + (size_t)(2 * l)     * WFRAG_PER_MAT,
            wfrag + (size_t)(2 * l + 1) * WFRAG_PER_MAT,
            b1 + (size_t)l * DIM, b2 + (size_t)l * DIM,
            node_cat + (size_t)l * DIM);
        h = node_cat + (size_t)l * DIM;
        hstride = OUT_STRIDE;
    }
    pool_kernel<<<((unsigned)N_NODES * (OUT_STRIDE / 4) + 255) / 256, 256, 0, stream>>>(
        node_cat, batch, graph_emb);
}
